// TACN_Influence_Model_86449101734285
// MI455X (gfx1250) — matrix operand order repaired and checked by reading, not compiled, not run on hardware
//
#include <hip/hip_runtime.h>
#include <hip/hip_bf16.h>

// ---- problem constants (match reference) ----
#define NUM_NODES 120000
#define TT 10
#define BB 64
#define RR 4
#define NN 64
#define CC 64
#define DD 128
#define II 64
#define KK 2
#define LL 8

typedef __attribute__((ext_vector_type(16))) _Float16 v16h;
typedef __attribute__((ext_vector_type(8)))  _Float16 v8h;
typedef __attribute__((ext_vector_type(8)))  float    v8f;

__device__ __forceinline__ v8f vzero8() {
  v8f a;
#pragma unroll
  for (int i = 0; i < 8; ++i) a[i] = 0.f;
  return a;
}

// Build a 16x32 f16 A-fragment (contiguous k order, same as B: lane m=l&15, half h=l>>4;
// halves [0..15] = A[m][kbase+16h .. +15])
// directly from fp32 global weights with arbitrary row/column strides.
__device__ __forceinline__ v16h load_a_g(const float* base, int mStride, int kStride,
                                         int m, int h, int kbase) {
  v16h a;
  const float* p = base + (size_t)m * mStride;
  const int c0 = kbase + 16 * h;
#pragma unroll
  for (int i = 0; i < 8; ++i) a[i] = (_Float16)p[(size_t)(c0 + i) * kStride];
  const int c1 = kbase + 16 * h + 8;
#pragma unroll
  for (int i = 0; i < 8; ++i) a[8 + i] = (_Float16)p[(size_t)(c1 + i) * kStride];
  return a;
}

// GEMM K-loop over one conv tap: B fragments streamed from LDS ([m][CIN] layout,
// so a fragment is 32 contiguous bytes per lane), A fragments pre-hoisted in regs.
template <int CIN>
__device__ __forceinline__ v8f conv_frags(const v16h (&afr)[CIN / 32],
                                          const _Float16* act_t, int lane, v8f acc) {
  const int n = lane & 15;
  const int h = lane >> 4;
#pragma unroll
  for (int kc = 0; kc < CIN / 32; ++kc) {
    v16h bfr = *(const v16h*)(act_t + n * CIN + kc * 32 + 16 * h);
    acc = __builtin_amdgcn_wmma_f32_16x16x32_f16(false, afr[kc], false, bfr,
                                                 (short)0, acc, false, false);
  }
  return acc;
}

// Store a 16x16 f32 C/D tile (lane n=l&15, VGPR v -> row cobase+v+8*(l>>4))
// as f16 into an LDS activation buffer laid out [m][64].
__device__ __forceinline__ void store_act(_Float16* base, int lane, int cobase, v8f v) {
  _Float16* p = base + (lane & 15) * CC + cobase + 8 * (lane >> 4);
  v8h o;
#pragma unroll
  for (int i = 0; i < 8; ++i) o[i] = (_Float16)v[i];
  *(v8h*)p = o;
}

__global__ void zero_out_kernel(float* out, int n) {
  int i = blockIdx.x * blockDim.x + threadIdx.x;
  if (i < n) out[i] = 0.f;
}

__global__ __launch_bounds__(128) void tacn_influence_kernel(
    const float* __restrict__ emb,    const float* __restrict__ w1_0,
    const float* __restrict__ b1_0,   const float* __restrict__ w2_0,
    const float* __restrict__ b2_0,   const float* __restrict__ down_w,
    const float* __restrict__ down_b, const float* __restrict__ w1,
    const float* __restrict__ b1,     const float* __restrict__ w2,
    const float* __restrict__ b2,     const float* __restrict__ rel_w,
    const int* __restrict__ align_,   const int* __restrict__ nidx,
    const unsigned char* __restrict__ nmask, float* __restrict__ out) {
  // LDS: rolling 2-slot level-0 input + full-T intermediate/state buffers. 48 KB.
  __shared__ alignas(32) _Float16 Xw[2 * 16 * DD];
  __shared__ alignas(32) _Float16 Yb[TT * 16 * CC];
  __shared__ alignas(32) _Float16 Hb[TT * 16 * CC];

  const int tid    = threadIdx.x;
  const int lane   = tid & 31;
  const int wave   = tid >> 5;
  const int cobase = wave * 16;   // this wave's 16 output channels
  const int mA     = lane & 15;
  const int hA     = lane >> 4;

  const int wg    = blockIdx.x;             // grid = R*B*(N/16) = 1024
  const int ntile = wg & 3;
  const int b     = (wg >> 2) & 63;
  const int r     = wg >> 8;
  const int nbase = ntile * 16;

  // ---- hoist level-0 weight fragments (fp32 -> f16, L2-resident) ----
  v16h a10[4], a11[4];   // conv1 (Cin=128), taps 0/1
  {
    const float* wb = w1_0 + ((size_t)r * CC + cobase) * DD * KK;  // [C][D][K]
#pragma unroll
    for (int kc = 0; kc < 4; ++kc) {
      a10[kc] = load_a_g(wb + 0, DD * KK, KK, mA, hA, kc * 32);
      a11[kc] = load_a_g(wb + 1, DD * KK, KK, mA, hA, kc * 32);
    }
  }
  v16h a20[2], a21[2];   // conv2 (Cin=64), taps 0/1
  {
    const float* wb = w2_0 + ((size_t)r * CC + cobase) * CC * KK;
#pragma unroll
    for (int kc = 0; kc < 2; ++kc) {
      a20[kc] = load_a_g(wb + 0, CC * KK, KK, mA, hA, kc * 32);
      a21[kc] = load_a_g(wb + 1, CC * KK, KK, mA, hA, kc * 32);
    }
  }
  v16h adn[4];           // 1x1 downsample (Cin=128)
  {
    const float* wb = down_w + ((size_t)r * CC + cobase) * DD;
#pragma unroll
    for (int kc = 0; kc < 4; ++kc)
      adn[kc] = load_a_g(wb, DD, 1, mA, hA, kc * 32);
  }
  float bv1[8], bv2[8], bvd[8];
#pragma unroll
  for (int v = 0; v < 8; ++v) {
    const int co = cobase + v + 8 * hA;
    bv1[v] = b1_0[r * CC + co];
    bv2[v] = b2_0[r * CC + co];
    bvd[v] = down_b[r * CC + co];
  }

  // gather assignment: 8 threads per neighbor, 16 channels each
  const int gm   = tid >> 3;
  const int gs   = tid & 7;
  const int gOff = (b * RR + r) * NN + nbase + gm;
  const int ent  = nidx[gOff];
  const bool msk = nmask[gOff] != 0;

  // ---- level 0: fused gather + conv1 + conv2 + downsample residual ----
  for (int t = 0; t < TT; ++t) {
    __syncthreads();              // protects X[(t-1)&1] read of previous iter
    {
      const int row = align_[(size_t)ent * TT + t];
      const bool ok = msk && (row >= 0);
      const int rr  = row < 0 ? 0 : row;
      const float* src = emb + ((size_t)t * NUM_NODES + rr) * DD + gs * 16;
      _Float16* dst = Xw + ((t & 1) * 16 + gm) * DD + gs * 16;
#pragma unroll
      for (int i = 0; i < 16; i += 4) {
        float4 vv = ok ? *(const float4*)(src + i) : make_float4(0.f, 0.f, 0.f, 0.f);
        dst[i + 0] = (_Float16)vv.x; dst[i + 1] = (_Float16)vv.y;
        dst[i + 2] = (_Float16)vv.z; dst[i + 3] = (_Float16)vv.w;
      }
    }
    __syncthreads();
    {   // Y[t] = relu(conv1(X[t-1],X[t]) + b1_0)
      v8f acc = vzero8();
      acc = conv_frags<DD>(a11, Xw + (t & 1) * 16 * DD, lane, acc);
      if (t >= 1) acc = conv_frags<DD>(a10, Xw + ((t & 1) ^ 1) * 16 * DD, lane, acc);
#pragma unroll
      for (int v = 0; v < 8; ++v) acc[v] = fmaxf(acc[v] + bv1[v], 0.f);
      store_act(Yb + t * 16 * CC, lane, cobase, acc);
    }
    __syncthreads();
    {   // H[t] = relu(relu(conv2(Y[t-1],Y[t]) + b2_0) + down(X[t]) + down_b)
      v8f acc = vzero8();
      acc = conv_frags<CC>(a21, Yb + t * 16 * CC, lane, acc);
      if (t >= 1) acc = conv_frags<CC>(a20, Yb + (t - 1) * 16 * CC, lane, acc);
      v8f ad = vzero8();
      ad = conv_frags<DD>(adn, Xw + (t & 1) * 16 * DD, lane, ad);
#pragma unroll
      for (int v = 0; v < 8; ++v) {
        const float y2 = fmaxf(acc[v] + bv2[v], 0.f);
        acc[v] = fmaxf(y2 + ad[v] + bvd[v], 0.f);
      }
      store_act(Hb + t * 16 * CC, lane, cobase, acc);
    }
  }

  // ---- levels 1..7: dilated residual blocks, all GEMMs on H/Y in LDS ----
  for (int lev = 1; lev < LL; ++lev) {
    const int d = 1 << lev;
    v16h c10[2], c11[2], c20[2], c21[2];
    {
      const float* wb = w1 + ((size_t)(r * (LL - 1) + (lev - 1)) * CC + cobase) * CC * KK;
#pragma unroll
      for (int kc = 0; kc < 2; ++kc) {
        c10[kc] = load_a_g(wb + 0, CC * KK, KK, mA, hA, kc * 32);
        c11[kc] = load_a_g(wb + 1, CC * KK, KK, mA, hA, kc * 32);
      }
    }
    {
      const float* wb = w2 + ((size_t)(r * (LL - 1) + (lev - 1)) * CC + cobase) * CC * KK;
#pragma unroll
      for (int kc = 0; kc < 2; ++kc) {
        c20[kc] = load_a_g(wb + 0, CC * KK, KK, mA, hA, kc * 32);
        c21[kc] = load_a_g(wb + 1, CC * KK, KK, mA, hA, kc * 32);
      }
    }
    float e1[8], e2[8];
#pragma unroll
    for (int v = 0; v < 8; ++v) {
      const size_t lb = (size_t)(r * (LL - 1) + (lev - 1)) * CC + cobase + v + 8 * hA;
      e1[v] = b1[lb];
      e2[v] = b2[lb];
    }

    __syncthreads();              // H complete from previous stage
    for (int t = 0; t < TT; ++t) {   // Y[t] = relu(conv1(H) + b1)
      v8f acc = vzero8();
      acc = conv_frags<CC>(c11, Hb + t * 16 * CC, lane, acc);
      if (t >= d) acc = conv_frags<CC>(c10, Hb + (t - d) * 16 * CC, lane, acc);
#pragma unroll
      for (int v = 0; v < 8; ++v) acc[v] = fmaxf(acc[v] + e1[v], 0.f);
      store_act(Yb + t * 16 * CC, lane, cobase, acc);
    }
    __syncthreads();
    for (int t = 0; t < TT; ++t) {   // H[t] = relu(relu(conv2(Y)+b2) + H[t])
      v8f acc = vzero8();
      acc = conv_frags<CC>(c21, Yb + t * 16 * CC, lane, acc);
      if (t >= d) acc = conv_frags<CC>(c20, Yb + (t - d) * 16 * CC, lane, acc);
      _Float16* hp = Hb + t * 16 * CC + (lane & 15) * CC + cobase + 8 * hA;
      const v8h hold = *(const v8h*)hp;
      v8h hnew;
#pragma unroll
      for (int v = 0; v < 8; ++v) {
        const float tmp = fmaxf(acc[v] + e2[v], 0.f);
        hnew[v] = (_Float16)fmaxf(tmp + (float)hold[v], 0.f);
      }
      *(v8h*)hp = hnew;
    }
  }

  // ---- influence GEMM (A = rel_w^T) + masked neighbor/relation reduction ----
  __syncthreads();
  v16h ar[2];
  {
    const float* wb = rel_w + (size_t)r * CC * II + cobase;   // A[i][c] = rel_w[c][i]
#pragma unroll
    for (int kc = 0; kc < 2; ++kc)
      ar[kc] = load_a_g(wb, 1, II, mA, hA, kc * 32);
  }
  v8f acc = vzero8();
  acc = conv_frags<CC>(ar, Hb + (TT - 1) * 16 * CC, lane, acc);
  const float mv = nmask[(b * RR + r) * NN + nbase + mA] ? 1.f : 0.f;
#pragma unroll
  for (int v = 0; v < 8; ++v) acc[v] *= mv;
#pragma unroll
  for (int off = 1; off < 16; off <<= 1) {   // butterfly over the 16 neighbor lanes
#pragma unroll
    for (int v = 0; v < 8; ++v) acc[v] += __shfl_xor(acc[v], off, 32);
  }
  if (mA == 0) {
#pragma unroll
    for (int v = 0; v < 8; ++v)
      atomicAdd(&out[b * II + cobase + v + 8 * hA], acc[v]);
  }
}

extern "C" void kernel_launch(void* const* d_in, const int* in_sizes, int n_in,
                              void* d_out, int out_size, void* d_ws, size_t ws_size,
                              hipStream_t stream) {
  (void)in_sizes; (void)n_in; (void)d_ws; (void)ws_size;
  const float* emb    = (const float*)d_in[0];
  const float* w1_0   = (const float*)d_in[1];
  const float* b1_0   = (const float*)d_in[2];
  const float* w2_0   = (const float*)d_in[3];
  const float* b2_0   = (const float*)d_in[4];
  const float* down_w = (const float*)d_in[5];
  const float* down_b = (const float*)d_in[6];
  const float* w1     = (const float*)d_in[7];
  const float* b1     = (const float*)d_in[8];
  const float* w2     = (const float*)d_in[9];
  const float* b2     = (const float*)d_in[10];
  const float* rel_w  = (const float*)d_in[11];
  const int* align_   = (const int*)d_in[12];
  const int* nidx     = (const int*)d_in[13];
  const unsigned char* nmask = (const unsigned char*)d_in[14];  // numpy bool = 1 byte
  float* out = (float*)d_out;

  zero_out_kernel<<<(out_size + 255) / 256, 256, 0, stream>>>(out, out_size);
  tacn_influence_kernel<<<RR * BB * (NN / 16), 128, 0, stream>>>(
      emb, w1_0, b1_0, w2_0, b2_0, down_w, down_b, w1, b1, w2, b2, rel_w,
      align_, nidx, nmask, out);
}
